// AttentionBlock1D_11682311045236
// MI455X (gfx1250) — matrix operand order repaired and checked by reading, not compiled, not run on hardware
//
#include <hip/hip_runtime.h>

typedef __attribute__((ext_vector_type(16))) _Float16 v16h;
typedef __attribute__((ext_vector_type(8)))  _Float16 h8;
typedef __attribute__((ext_vector_type(8)))  float    v8f;

#define B_ 8
#define C_ 512
#define S_ 1024
#define H_ 8
#define D_ 64

static __device__ __forceinline__ v16h cat16(h8 lo, h8 hi) {
  return __builtin_shufflevector(lo, hi, 0,1,2,3,4,5,6,7,8,9,10,11,12,13,14,15);
}
static __device__ __forceinline__ v8f wmma16(v16h a, v16h b, v8f c) {
  return __builtin_amdgcn_wmma_f32_16x16x32_f16(false, a, false, b, (short)0, c, false, false);
}
static __device__ __forceinline__ v8f vzero8() {
  v8f z = {0.f,0.f,0.f,0.f,0.f,0.f,0.f,0.f};
  return z;
}

// CDNA5 async global->LDS copy (16B per lane), tracked by ASYNCcnt.
static __device__ __forceinline__ void async_copy16(const void* gptr, void* lptr) {
  uint32_t l = (uint32_t)(size_t)lptr;      // low 32 bits of generic addr = DS offset
  uint64_t g = (uint64_t)(size_t)gptr;
  asm volatile("global_load_async_to_lds_b128 %0, %1, off" : : "v"(l), "v"(g) : "memory");
}
static __device__ __forceinline__ void async_wait0() {
  asm volatile("s_wait_asynccnt 0" : : : "memory");
}

// Stage one 128x32 A tile + 128x32 B tile (K-contiguous rows, stride C_).
static __device__ __forceinline__ void stage_tiles(
    const _Float16* __restrict__ gA, const _Float16* __restrict__ gB,
    _Float16 (*lA)[32], _Float16 (*lB)[32], int t) {
#pragma unroll
  for (int r = 0; r < 2; ++r) {
    int ci = t * 2 + r;
    int row = ci >> 2, kc = (ci & 3) * 8;
    async_copy16(gA + (size_t)row * C_ + kc, &lA[row][kc]);
    async_copy16(gB + (size_t)row * C_ + kc, &lB[row][kc]);
  }
}

// ---------------------------------------------------------------- convert
__global__ void cvt_f32_f16_kernel(const float* __restrict__ src,
                                   _Float16* __restrict__ dst, int n) {
  int i = blockIdx.x * blockDim.x + threadIdx.x;
  if (i < n) dst[i] = (_Float16)src[i];
}

// ---------------------------------------------------------------- groupnorm
// one block per (b, g). Writes xnT in [B, S, C] layout (position-major) so the
// QKV GEMM's B operand is contiguous along K (= channels).
__global__ __launch_bounds__(256) void groupnorm_kernel(
    const float* __restrict__ x, const float* __restrict__ gamma,
    const float* __restrict__ beta, _Float16* __restrict__ xnT) {
  __shared__ float red[256];
  __shared__ float stats[2];
  const int bg = blockIdx.x;
  const int b = bg >> 3, g = bg & 7;
  const float* base = x + ((size_t)(b * C_ + g * 64)) * S_;
  const int t = threadIdx.x;

  float s = 0.f, ss = 0.f;
  for (int i = t; i < 65536 / 4; i += 256) {
    float4 v = ((const float4*)base)[i];
    s  += v.x + v.y + v.z + v.w;
    ss += v.x * v.x + v.y * v.y + v.z * v.z + v.w * v.w;
  }
  red[t] = s; __syncthreads();
  for (int o = 128; o > 0; o >>= 1) { if (t < o) red[t] += red[t + o]; __syncthreads(); }
  if (t == 0) stats[0] = red[0] * (1.0f / 65536.0f);
  __syncthreads();
  red[t] = ss; __syncthreads();
  for (int o = 128; o > 0; o >>= 1) { if (t < o) red[t] += red[t + o]; __syncthreads(); }
  if (t == 0) {
    float m = stats[0];
    float var = red[0] * (1.0f / 65536.0f) - m * m;
    stats[1] = rsqrtf(var + 1e-5f);
  }
  __syncthreads();
  const float mean = stats[0], inv = stats[1];

  _Float16* dstb = xnT + ((size_t)b * S_) * C_;   // [S][C]
  for (int it = 0; it < 32; ++it) {
    int idx = t + 256 * it;          // 0..8191
    int s2 = idx & 1023;             // position
    int cc8 = idx >> 10;             // which 8-channel chunk of the group
    h8 o;
#pragma unroll
    for (int e = 0; e < 8; ++e) {
      int cl = cc8 * 8 + e;          // channel within group
      float v = base[(size_t)cl * S_ + s2];
      int c = g * 64 + cl;
      o[e] = (_Float16)((v - mean) * inv * gamma[c] + beta[c]);
    }
    *(h8*)(dstb + (size_t)s2 * C_ + g * 64 + cc8 * 8) = o;
  }
}

// ---------------------------------------------------------------- qkv gemm
// qkv[o][s] = sum_c W[o][c] * xnT[b][s][c];  M=1536, K=512, N=1024 per batch.
// Double-buffered async-to-LDS staging; A/B tiles both K-contiguous.
// scatter: Q -> qt [B,H,D,S], K -> kk [B,H,S,D], V -> vt [B,H,D,S]  (f16)
__global__ __launch_bounds__(256) void qkv_gemm_kernel(
    const _Float16* __restrict__ W, const _Float16* __restrict__ XT,
    _Float16* __restrict__ qt, _Float16* __restrict__ kk,
    _Float16* __restrict__ vt) {
  __shared__ _Float16 As[2][128][32];
  __shared__ _Float16 Bs[2][128][32];
  const int b = blockIdx.z;
  const int mBase = blockIdx.y * 128;
  const int nBase = blockIdx.x * 128;
  const int t = threadIdx.x;
  const int l = t & 31, w = t >> 5;
  const int wm = w & 3, wn = w >> 2;
  const int ln = l & 15, lh = l >> 4;
  const _Float16* Ag = W + (size_t)mBase * C_;
  const _Float16* Bg = XT + (size_t)b * S_ * C_ + (size_t)nBase * C_;

  v8f acc[2][4];
#pragma unroll
  for (int mi = 0; mi < 2; ++mi)
#pragma unroll
    for (int ni = 0; ni < 4; ++ni) acc[mi][ni] = vzero8();

  stage_tiles(Ag, Bg, As[0], Bs[0], t);
  async_wait0();
  __syncthreads();

  for (int it = 0; it < C_ / 32; ++it) {
    const int k0 = it * 32;
    const int cur = it & 1;
    if (it + 1 < C_ / 32)
      stage_tiles(Ag + k0 + 32, Bg + k0 + 32, As[cur ^ 1], Bs[cur ^ 1], t);
    if (it + 2 < C_ / 32) {
      __builtin_prefetch((const void*)(Ag + (size_t)(t >> 1) * C_ + k0 + 64), 0, 1);
      __builtin_prefetch((const void*)(Bg + (size_t)(t >> 1) * C_ + k0 + 64), 0, 1);
    }

    v16h af[2], bf[4];
#pragma unroll
    for (int mi = 0; mi < 2; ++mi) {
      int row = wm * 32 + mi * 16 + ln;
      af[mi] = cat16(*(const h8*)&As[cur][row][lh * 8],
                     *(const h8*)&As[cur][row][16 + lh * 8]);
    }
#pragma unroll
    for (int ni = 0; ni < 4; ++ni) {
      int nrow = wn * 64 + ni * 16 + ln;
      bf[ni] = cat16(*(const h8*)&Bs[cur][nrow][lh * 8],
                     *(const h8*)&Bs[cur][nrow][16 + lh * 8]);
    }
#pragma unroll
    for (int mi = 0; mi < 2; ++mi)
#pragma unroll
      for (int ni = 0; ni < 4; ++ni)
        acc[mi][ni] = wmma16(af[mi], bf[ni], acc[mi][ni]);

    async_wait0();
    __syncthreads();
  }

  // scatter (segment test is uniform per 16-row tile: boundaries are 512-aligned)
#pragma unroll
  for (int mi = 0; mi < 2; ++mi) {
#pragma unroll
    for (int ni = 0; ni < 4; ++ni) {
      const int oBase = mBase + wm * 32 + mi * 16;
      const int s = nBase + wn * 64 + ni * 16 + ln;
      if (oBase < C_) {                       // Q -> [B,H,D,S]
#pragma unroll
        for (int j = 0; j < 8; ++j) {
          int o = oBase + j + 8 * lh;
          int h = o >> 6, d = o & 63;
          qt[(((size_t)(b * H_ + h) * D_ + d) << 10) + s] = (_Float16)acc[mi][ni][j];
        }
      } else if (oBase < 2 * C_) {            // K -> [B,H,S,D]
#pragma unroll
        for (int j = 0; j < 8; ++j) {
          int c = oBase - C_ + j + 8 * lh;
          int h = c >> 6, d = c & 63;
          kk[((((size_t)(b * H_ + h) << 10) + s) * D_) + d] = (_Float16)acc[mi][ni][j];
        }
      } else {                                // V -> [B,H,D,S]
#pragma unroll
        for (int j = 0; j < 8; ++j) {
          int c = oBase - 2 * C_ + j + 8 * lh;
          int h = c >> 6, d = c & 63;
          vt[(((size_t)(b * H_ + h) * D_ + d) << 10) + s] = (_Float16)acc[mi][ni][j];
        }
      }
    }
  }
}

// ---------------------------------------------------------------- attention
// one block (4 waves) per (b, h, 16-query tile).
__global__ __launch_bounds__(128) void attn_kernel(
    const _Float16* __restrict__ qt, const _Float16* __restrict__ kk,
    const _Float16* __restrict__ vt, _Float16* __restrict__ aoT) {
  __shared__ _Float16 sc[16][1024];   // scores then P, f16
  __shared__ float red[16][8];
  __shared__ float rmax[16];
  __shared__ float rscale[16];

  const int blk = blockIdx.x;
  const int qt16 = blk & 63;
  const int h = (blk >> 6) & 7;
  const int b = blk >> 9;
  const int bh = b * H_ + h;
  const int t = threadIdx.x, l = t & 31, w = t >> 5;
  const int ln = l & 15, lh = l >> 4;
  const int qbase = qt16 * 16;
  const _Float16* Kb = kk + ((size_t)bh << 10) * D_;   // [S][D]
  const _Float16* Qt = qt + ((size_t)bh * D_ << 10);   // [D][S]
  const _Float16* Vt = vt + ((size_t)bh * D_ << 10);   // [D][S]

  // Q as B-matrix fragments (fixed per block): b[e] = Qt[ch*32 + lh*8 + (e&7) + (e>>3)*16][qcol]
  v16h bq[2];
#pragma unroll
  for (int ch = 0; ch < 2; ++ch) {
    v16h bb;
#pragma unroll
    for (int e = 0; e < 16; ++e)
      bb[e] = Qt[((size_t)(ch * 32 + lh * 8 + (e & 7) + (e >> 3) * 16) << 10) + qbase + ln];
    bq[ch] = bb;
  }

  // scores^T = K * Qt  (M = key pos, N = query) -> stored transposed sc[q][k]
  for (int kt = w; kt < 64; kt += 4) {
    int krow = kt * 16 + ln;
    v8f acc = vzero8();
#pragma unroll
    for (int ch = 0; ch < 2; ++ch) {
      const _Float16* arow = Kb + (size_t)krow * D_ + ch * 32 + lh * 8;
      v16h a = cat16(*(const h8*)arow, *(const h8*)(arow + 16));
      acc = wmma16(a, bq[ch], acc);
    }
#pragma unroll
    for (int j = 0; j < 8; ++j) {
      int kpos = kt * 16 + j + 8 * lh;
      sc[ln][kpos] = (_Float16)(acc[j] * 0.125f);
    }
  }
  __syncthreads();

  // softmax over k (1024) for each of 16 query rows; 8 threads per row
  const int q = t >> 3, seg = t & 7;
  {
    float m = -1e30f;
    for (int k2 = seg * 128; k2 < seg * 128 + 128; ++k2)
      m = fmaxf(m, (float)sc[q][k2]);
    red[q][seg] = m;
  }
  __syncthreads();
  if (seg == 0) {
    float m = red[q][0];
#pragma unroll
    for (int e = 1; e < 8; ++e) m = fmaxf(m, red[q][e]);
    rmax[q] = m;
  }
  __syncthreads();
  {
    float m = rmax[q], s = 0.f;
    for (int k2 = seg * 128; k2 < seg * 128 + 128; ++k2) {
      float e = __expf((float)sc[q][k2] - m);
      sc[q][k2] = (_Float16)e;
      s += e;
    }
    red[q][seg] = s;
  }
  __syncthreads();
  if (seg == 0) {
    float s = 0.f;
#pragma unroll
    for (int e = 0; e < 8; ++e) s += red[q][e];
    rscale[q] = 1.0f / s;
  }
  __syncthreads();

  // O = P(16x1024) x V(1024x64); wave w owns d-tile [w*16, w*16+16)
  v8f o = vzero8();
  const int dcol = w * 16 + ln;
  for (int kc = 0; kc < 32; ++kc) {
    const _Float16* prow = &sc[ln][kc * 32 + lh * 8];
    v16h a = cat16(*(const h8*)prow, *(const h8*)(prow + 16));
    const _Float16* vrow = Vt + ((size_t)dcol << 10) + kc * 32 + lh * 8;
    v16h bb = cat16(*(const h8*)vrow, *(const h8*)(vrow + 16));
    o = wmma16(a, bb, o);
  }
  // scale by 1/rowsum and store transposed: aoT[B, S, C], c = h*64 + d
  const int cidx = h * D_ + dcol;
#pragma unroll
  for (int j = 0; j < 8; ++j) {
    int s_pos = qbase + j + 8 * lh;
    float val = o[j] * rscale[j + 8 * lh];
    aoT[((size_t)(b * S_ + s_pos)) * C_ + cidx] = (_Float16)val;
  }
}

// ---------------------------------------------------------------- out proj
// out[b][o][s] = sum_c Wout[o][c]*aoT[b][s][c] + bias[o] + x[b][o][s]
__global__ __launch_bounds__(256) void out_gemm_kernel(
    const _Float16* __restrict__ W, const _Float16* __restrict__ AOT,
    const float* __restrict__ bias, const float* __restrict__ xin,
    float* __restrict__ out) {
  __shared__ _Float16 As[2][128][32];
  __shared__ _Float16 Bs[2][128][32];
  const int b = blockIdx.z;
  const int mBase = blockIdx.y * 128;
  const int nBase = blockIdx.x * 128;
  const int t = threadIdx.x;
  const int l = t & 31, w = t >> 5;
  const int wm = w & 3, wn = w >> 2;
  const int ln = l & 15, lh = l >> 4;
  const _Float16* Ag = W + (size_t)mBase * C_;
  const _Float16* Bg = AOT + (size_t)b * S_ * C_ + (size_t)nBase * C_;

  v8f acc[2][4];
#pragma unroll
  for (int mi = 0; mi < 2; ++mi)
#pragma unroll
    for (int ni = 0; ni < 4; ++ni) acc[mi][ni] = vzero8();

  stage_tiles(Ag, Bg, As[0], Bs[0], t);
  async_wait0();
  __syncthreads();

  for (int it = 0; it < C_ / 32; ++it) {
    const int k0 = it * 32;
    const int cur = it & 1;
    if (it + 1 < C_ / 32)
      stage_tiles(Ag + k0 + 32, Bg + k0 + 32, As[cur ^ 1], Bs[cur ^ 1], t);
    if (it + 2 < C_ / 32) {
      __builtin_prefetch((const void*)(Ag + (size_t)(t >> 1) * C_ + k0 + 64), 0, 1);
      __builtin_prefetch((const void*)(Bg + (size_t)(t >> 1) * C_ + k0 + 64), 0, 1);
    }

    v16h af[2], bf[4];
#pragma unroll
    for (int mi = 0; mi < 2; ++mi) {
      int row = wm * 32 + mi * 16 + ln;
      af[mi] = cat16(*(const h8*)&As[cur][row][lh * 8],
                     *(const h8*)&As[cur][row][16 + lh * 8]);
    }
#pragma unroll
    for (int ni = 0; ni < 4; ++ni) {
      int nrow = wn * 64 + ni * 16 + ln;
      bf[ni] = cat16(*(const h8*)&Bs[cur][nrow][lh * 8],
                     *(const h8*)&Bs[cur][nrow][16 + lh * 8]);
    }
#pragma unroll
    for (int mi = 0; mi < 2; ++mi)
#pragma unroll
      for (int ni = 0; ni < 4; ++ni)
        acc[mi][ni] = wmma16(af[mi], bf[ni], acc[mi][ni]);

    async_wait0();
    __syncthreads();
  }

#pragma unroll
  for (int mi = 0; mi < 2; ++mi) {
#pragma unroll
    for (int ni = 0; ni < 4; ++ni) {
      int oBase = mBase + wm * 32 + mi * 16;
      int s = nBase + wn * 64 + ni * 16 + ln;
#pragma unroll
      for (int j = 0; j < 8; ++j) {
        int o = oBase + j + 8 * lh;
        size_t idx = ((size_t)(b * C_ + o) << 10) + s;
        out[idx] = acc[mi][ni][j] + bias[o] + xin[idx];
      }
    }
  }
}

// ---------------------------------------------------------------- launch
extern "C" void kernel_launch(void* const* d_in, const int* in_sizes, int n_in,
                              void* d_out, int out_size, void* d_ws, size_t ws_size,
                              hipStream_t stream) {
  (void)in_sizes; (void)n_in; (void)out_size; (void)ws_size;
  const float* x     = (const float*)d_in[0];
  const float* gamma = (const float*)d_in[1];
  const float* beta  = (const float*)d_in[2];
  const float* wqkv  = (const float*)d_in[3];
  const float* wout  = (const float*)d_in[4];
  const float* bout  = (const float*)d_in[5];
  float* out = (float*)d_out;

  char* ws = (char*)d_ws;
  size_t off = 0;
  auto alloc = [&](size_t bytes) {
    void* p = ws + off;
    off += (bytes + 255) & ~(size_t)255;
    return p;
  };
  _Float16* wqkv_h = (_Float16*)alloc((size_t)3 * C_ * C_ * 2);
  _Float16* wout_h = (_Float16*)alloc((size_t)C_ * C_ * 2);
  _Float16* xnT_h  = (_Float16*)alloc((size_t)B_ * S_ * C_ * 2);
  _Float16* qt_h   = (_Float16*)alloc((size_t)B_ * H_ * D_ * S_ * 2);
  _Float16* k_h    = (_Float16*)alloc((size_t)B_ * H_ * S_ * D_ * 2);
  _Float16* vt_h   = (_Float16*)alloc((size_t)B_ * H_ * D_ * S_ * 2);
  _Float16* aoT_h  = (_Float16*)alloc((size_t)B_ * S_ * C_ * 2);

  cvt_f32_f16_kernel<<<(3 * C_ * C_ + 255) / 256, 256, 0, stream>>>(wqkv, wqkv_h, 3 * C_ * C_);
  cvt_f32_f16_kernel<<<(C_ * C_ + 255) / 256, 256, 0, stream>>>(wout, wout_h, C_ * C_);

  groupnorm_kernel<<<B_ * 8, 256, 0, stream>>>(x, gamma, beta, xnT_h);

  qkv_gemm_kernel<<<dim3(S_ / 128, (3 * C_) / 128, B_), 256, 0, stream>>>(
      wqkv_h, xnT_h, qt_h, k_h, vt_h);

  attn_kernel<<<B_ * H_ * (S_ / 16), 128, 0, stream>>>(qt_h, k_h, vt_h, aoT_h);

  out_gemm_kernel<<<dim3(S_ / 128, C_ / 128, B_), 256, 0, stream>>>(
      wout_h, aoT_h, bout, x, out);
}
